// PointNetSetAbstractionMsg_49898930045498
// MI455X (gfx1250) — matrix operand order repaired and checked by reading, not compiled, not run on hardware
//
#include <hip/hip_runtime.h>
#include <hip/hip_bf16.h>
#include <stdint.h>

// ---------------------------------------------------------------------------
// PointNet++ Set Abstraction MSG for MI455X (gfx1250, wave32, WMMA bf16)
// ---------------------------------------------------------------------------

typedef __bf16 bf16;
typedef __attribute__((ext_vector_type(16))) __bf16 v16bf;
typedef __attribute__((ext_vector_type(8)))  __bf16 v8bf;
typedef __attribute__((ext_vector_type(8)))  float  v8f;

#define BATCH   8
#define NPTS    16384
#define SPTS    1024     // NPOINT
#define OUTCH   640
#define BN_EPS  1e-5f

// ===========================================================================
// 1. Farthest point sampling: one workgroup per batch.
//    Entire batch state lives in LDS (CDNA5: 320KB/WGP):
//      xs/ys/zs: 3*16384 f32 = 192KB, dist: 64KB, reduce: 8KB  -> 264KB
// ===========================================================================
__global__ __launch_bounds__(1024, 1)
void fps_kernel(const float* __restrict__ xyz, int* __restrict__ fps_idx) {
    extern __shared__ float smem[];
    float* xs   = smem;
    float* ys   = xs + NPTS;
    float* zs   = ys + NPTS;
    float* dist = zs + NPTS;
    float* rd   = dist + NPTS;          // 1024 floats
    int*   ri   = (int*)(rd + 1024);    // 1024 ints

    const int b = blockIdx.x;
    const int t = threadIdx.x;
    const float* px = xyz + (size_t)b * 3 * NPTS;

    for (int n = t; n < NPTS; n += 1024) {
        xs[n] = px[n];
        ys[n] = px[NPTS + n];
        zs[n] = px[2 * NPTS + n];
        dist[n] = 1e10f;
    }
    __syncthreads();

    int far = 0;
    for (int it = 0; it < SPTS; ++it) {
        if (t == 0) fps_idx[b * SPTS + it] = far;   // scan emits carry BEFORE update
        const float cx = xs[far], cy = ys[far], cz = zs[far];

        float best = -1.0f;
        int   bi   = 0;
        for (int n = t; n < NPTS; n += 1024) {
            float dx = xs[n] - cx, dy = ys[n] - cy, dz = zs[n] - cz;
            float d  = dx * dx + dy * dy + dz * dz;
            float dd = dist[n];
            if (d < dd) { dd = d; dist[n] = dd; }
            if (dd > best) { best = dd; bi = n; }   // strict > keeps lowest index
        }
        rd[t] = best; ri[t] = bi;
        __syncthreads();
        for (int s = 512; s > 0; s >>= 1) {
            if (t < s) {
                float ov = rd[t + s]; int oi = ri[t + s];
                if (ov > rd[t] || (ov == rd[t] && oi < ri[t])) { rd[t] = ov; ri[t] = oi; }
            }
            __syncthreads();
        }
        far = ri[0];
        __syncthreads();
    }
}

// Gather sampled centers: ws layout (B,S,3) and d_out layout (B,3,S)
__global__ void gather_newxyz_kernel(const float* __restrict__ xyz,
                                     const int* __restrict__ fps_idx,
                                     float* __restrict__ newxyz_ws,
                                     float* __restrict__ out_xyz) {
    int tid = blockIdx.x * blockDim.x + threadIdx.x;
    if (tid >= BATCH * SPTS * 3) return;
    int c = tid % 3;
    int s = (tid / 3) % SPTS;
    int b = tid / (3 * SPTS);
    int idx = fps_idx[b * SPTS + s];
    float v = xyz[((size_t)b * 3 + c) * NPTS + idx];
    newxyz_ws[((size_t)b * SPTS + s) * 3 + c] = v;
    out_xyz[((size_t)b * 3 + c) * SPTS + s]   = v;
}

// ===========================================================================
// 2. Ball query: one wave per query point. Ordered scan, ballot compaction.
//    Equivalent to reference's where/sort/take-first-K (indices ascending).
// ===========================================================================
__global__ __launch_bounds__(256)
void ball_query_kernel(const float* __restrict__ xyz,
                       const float* __restrict__ newxyz,
                       int* __restrict__ gidx, float r2, int K) {
    int gtid = blockIdx.x * blockDim.x + threadIdx.x;
    int wave = gtid >> 5;
    int lane = gtid & 31;
    if (wave >= BATCH * SPTS) return;
    int b = wave / SPTS, s = wave % SPTS;

    const float qx = newxyz[((size_t)b * SPTS + s) * 3 + 0];
    const float qy = newxyz[((size_t)b * SPTS + s) * 3 + 1];
    const float qz = newxyz[((size_t)b * SPTS + s) * 3 + 2];
    const float* px = xyz + (size_t)b * 3 * NPTS;
    int* out = gidx + (size_t)wave * K;

    int found = 0;
    int firstN = -1;
    for (int base = 0; base < NPTS; base += 32) {
        int n = base + lane;
        float dx = px[n] - qx;
        float dy = px[NPTS + n] - qy;
        float dz = px[2 * NPTS + n] - qz;
        bool ok = (dx * dx + dy * dy + dz * dz) <= r2;
        unsigned mask = (unsigned)__ballot(ok);
        if (firstN < 0 && mask) firstN = base + __ffs((int)mask) - 1;
        int before = __popc(mask & ((1u << lane) - 1u));
        int slot = found + before;
        if (ok && slot < K) out[slot] = n;
        found += __popc(mask);
        if (found >= K) break;
    }
    if (firstN < 0) firstN = 0;
    for (int s2 = found + lane; s2 < K; s2 += 32) out[s2] = firstN;  // pad with first
}

// ===========================================================================
// 3. Grouping: build layer-0 input rows [dx,dy,dz, pts(61)] = 64ch, bf16.
// ===========================================================================
__global__ void group_kernel(const float* __restrict__ xyz,
                             const float* __restrict__ pts,
                             const int* __restrict__ gidx,
                             const float* __restrict__ newxyz,
                             bf16* __restrict__ X, int Rows, int K) {
    int tid = blockIdx.x * blockDim.x + threadIdx.x;
    if (tid >= Rows * 64) return;
    int r = tid >> 6, c = tid & 63;
    int b = r / (SPTS * K);
    int s = (r % (SPTS * K)) / K;
    int idx = gidx[r];
    float v;
    if (c < 3)
        v = xyz[((size_t)b * 3 + c) * NPTS + idx] -
            newxyz[((size_t)b * SPTS + s) * 3 + c];
    else
        v = pts[((size_t)b * 61 + (c - 3)) * NPTS + idx];
    X[(size_t)r * 64 + c] = (bf16)v;
}

// ===========================================================================
// 4. Weight pre-swizzle into the B-fragment order this GEMM uses (contiguous k, same as A):
//    frag[((oT*KC + kc)*32 + lane)*16 + e] =
//        W[oT*16 + (lane&15)][kc*32 + (lane>>4)*16 + e]
// ===========================================================================
__global__ void wprep_kernel(const float* __restrict__ W, bf16* __restrict__ Wf,
                             int Cin, int Cout) {
    int tid = blockIdx.x * blockDim.x + threadIdx.x;
    if (tid >= Cin * Cout) return;
    int e    = tid & 15;
    int lane = (tid >> 4) & 31;
    int KC   = Cin >> 5;
    int kc   = (tid >> 9) % KC;
    int oT   = tid / (512 * KC);
    int n = lane & 15;
    int k = kc * 32 + (lane >> 4) * 16 + e;
    Wf[tid] = (bf16)W[(size_t)(oT * 16 + n) * Cin + k];
}

// ===========================================================================
// 5. GEMM: Y[Rows x Cout] = X[Rows x Cin] * W^T + bias  (bf16 in, f32 acc)
//    - one wave computes a 16(M) x 64(O) strip: A fragment loaded once per
//      32-deep k-chunk and fed to 4 WMMAs (4 accumulators)
//    - entire pre-swizzled weight array (<= 64KB) staged in LDS per block:
//      B-fragment reads become ds_load_b128 (DScnt), leaving vmem/LOADcnt
//      exclusively for the streaming A fragments.
// ===========================================================================
__global__ __launch_bounds__(256)
void gemm_bf16_kernel(const bf16* __restrict__ X, const bf16* __restrict__ Wf,
                      const float* __restrict__ bias, bf16* __restrict__ Y,
                      int Rows, int Cin, int Cout) {
    extern __shared__ bf16 wlds[];               // Cin*Cout bf16 (<= 64KB)
    const int nW = Cin * Cout;
    for (int i = threadIdx.x * 8; i < nW; i += 256 * 8)
        *(v8bf*)(wlds + i) = *(const v8bf*)(Wf + i);
    __syncthreads();

    int wave = (blockIdx.x * blockDim.x + threadIdx.x) >> 5;
    int lane = threadIdx.x & 31;
    int OG = Cout >> 6;                          // groups of 4 o-tiles (Cout % 64 == 0)
    int totalGroups = (Rows >> 4) * OG;
    if (wave >= totalGroups) return;
    int mT = wave / OG, oG = wave % OG;

    int n    = lane & 15;
    int half = lane >> 4;
    int KC   = Cin >> 5;

    v8f acc[4];
#pragma unroll
    for (int u = 0; u < 4; ++u) {
        float bv = bias[oG * 64 + u * 16 + n];
        acc[u] = (v8f){bv, bv, bv, bv, bv, bv, bv, bv};
    }

    const bf16* arow = X + (size_t)((mT << 4) + n) * Cin + half * 16;
    const bf16* wp   = wlds + (size_t)(oG * 4) * KC * 512 + lane * 16;

    for (int kc = 0; kc < KC; ++kc) {
        v8bf lo = *(const v8bf*)(arow);          // K = kb..kb+7
        v8bf hi = *(const v8bf*)(arow + 8);      // K = kb+8..kb+15
        v16bf a = __builtin_shufflevector(lo, hi,
                    0,1,2,3,4,5,6,7,8,9,10,11,12,13,14,15);
#pragma unroll
        for (int u = 0; u < 4; ++u) {
            v16bf bm = *(const v16bf*)(wp + (size_t)u * KC * 512);  // ds_load_b128 x2
            acc[u] = __builtin_amdgcn_wmma_f32_16x16x32_bf16(
                         false, a, false, bm, (short)0, acc[u], false, false);
        }
        arow += 32;
        wp   += 512;
    }

    int mBase = (mT << 4) + half * 8;
    bf16* yp = Y + (size_t)mBase * Cout + oG * 64 + n;
#pragma unroll
    for (int u = 0; u < 4; ++u) {
#pragma unroll
        for (int r = 0; r < 8; ++r)
            yp[(size_t)r * Cout + u * 16] = (bf16)acc[u][r];
    }
}

// ===========================================================================
// 6. BatchNorm statistics: per-channel sum / sumsq over all rows.
// ===========================================================================
__global__ void zero_kernel(float* __restrict__ p, int n) {
    int tid = blockIdx.x * blockDim.x + threadIdx.x;
    if (tid < n) p[tid] = 0.0f;
}

__global__ __launch_bounds__(256)
void stats_kernel(const bf16* __restrict__ Y, size_t total, int Cout,
                  float* __restrict__ sum, float* __restrict__ sq) {
    __shared__ float ls[256], lq[256];
    int t = threadIdx.x;
    int c = (int)((blockIdx.x * 256u + t) % (unsigned)Cout);  // invariant: 256 % Cout == 0
    float s = 0.0f, q = 0.0f;
    for (size_t i = blockIdx.x * 256u + t; i < total; i += (size_t)gridDim.x * 256u) {
        float v = (float)Y[i];
        s += v; q += v * v;
    }
    ls[t] = s; lq[t] = q;
    __syncthreads();
    for (int st = 128; st >= Cout; st >>= 1) {
        if (t < st) { ls[t] += ls[t + st]; lq[t] += lq[t + st]; }
        __syncthreads();
    }
    if (t < Cout) {
        atomicAdd(&sum[c], ls[t]);
        atomicAdd(&sq[c],  lq[t]);
    }
}

__global__ void bn_finalize_kernel(const float* __restrict__ sum,
                                   const float* __restrict__ sq,
                                   const float* __restrict__ g,
                                   const float* __restrict__ bt,
                                   float invN, float* __restrict__ a,
                                   float* __restrict__ c, int Cout) {
    int t = blockIdx.x * blockDim.x + threadIdx.x;
    if (t >= Cout) return;
    float mean = sum[t] * invN;
    float var  = sq[t] * invN - mean * mean;
    float av   = g[t] * rsqrtf(var + BN_EPS);
    a[t] = av;
    c[t] = bt[t] - av * mean;
}

// ===========================================================================
// 7. Fused BN+ReLU (feeds next layer) / BN+ReLU+max-over-K (final layer)
// ===========================================================================
__global__ void bn_relu_kernel(const bf16* __restrict__ Y,
                               const float* __restrict__ a,
                               const float* __restrict__ c,
                               bf16* __restrict__ Xn, size_t total, int Cout) {
    size_t tid = (size_t)blockIdx.x * blockDim.x + threadIdx.x;
    if (tid >= total) return;
    int ch = (int)(tid % (unsigned)Cout);
    float v = fmaf(a[ch], (float)Y[tid], c[ch]);
    Xn[tid] = (bf16)fmaxf(v, 0.0f);
}

__global__ void bn_relu_maxpool_kernel(const bf16* __restrict__ Y,
                                       const float* __restrict__ a,
                                       const float* __restrict__ c,
                                       float* __restrict__ out, int K, int Cout,
                                       int chanOff) {
    int tid = blockIdx.x * blockDim.x + threadIdx.x;
    if (tid >= BATCH * SPTS * Cout) return;
    int o = tid % Cout;
    int s = (tid / Cout) % SPTS;
    int b = tid / (Cout * SPTS);
    const float av = a[o], cv = c[o];
    const bf16* yp = Y + ((size_t)(b * SPTS + s) * K) * Cout + o;
    float m = -3.4e38f;
    for (int k = 0; k < K; ++k) {
        float v = fmaxf(fmaf(av, (float)yp[(size_t)k * Cout], cv), 0.0f);
        m = fmaxf(m, v);
    }
    out[(size_t)b * OUTCH * SPTS + (size_t)(chanOff + o) * SPTS + s] = m;
}

// ===========================================================================
// Host orchestration
// ===========================================================================
extern "C" void kernel_launch(void* const* d_in, const int* in_sizes, int n_in,
                              void* d_out, int out_size, void* d_ws, size_t ws_size,
                              hipStream_t stream) {
    (void)in_sizes; (void)n_in; (void)out_size; (void)ws_size;

    const float* xyz = (const float*)d_in[0];
    const float* pts = (const float*)d_in[1];

    static const int   MLPS[3][3] = {{64,64,128},{128,128,256},{128,128,256}};
    static const int   KS[3]      = {16, 32, 64};
    static const float R2[3]      = {0.01f, 0.04f, 0.16f};
    static const int   CHOFF[3]   = {0, 128, 384};

    // workspace carve (256B-aligned regions)
    char* w = (char*)d_ws;
    size_t off = 0;
    auto carve = [&](size_t bytes) { void* p = w + off; off += (bytes + 255) & ~(size_t)255; return p; };
    bf16*  Xbuf    = (bf16*) carve((size_t)524288 * 128 * 2);  // layer inputs (Cin <= 128)
    bf16*  Ybuf    = (bf16*) carve((size_t)524288 * 256 * 2);  // layer outputs (Cout <= 256)
    int*   gidx    = (int*)  carve((size_t)BATCH * SPTS * 64 * 4);
    int*   fpsIdx  = (int*)  carve((size_t)BATCH * SPTS * 4);
    float* newxyz  = (float*)carve((size_t)BATCH * SPTS * 3 * 4);
    bf16*  wfrag   = (bf16*) carve((size_t)128 * 256 * 2);
    float* bnSum   = (float*)carve(256 * 4);
    float* bnSq    = (float*)carve(256 * 4);
    float* bnA     = (float*)carve(256 * 4);
    float* bnC     = (float*)carve(256 * 4);

    float* outXyz = (float*)d_out;                       // (B,3,S)
    float* outPts = outXyz + (size_t)BATCH * 3 * SPTS;   // (B,640,S)

    // ---- FPS + center gather -------------------------------------------
    size_t fpsSmem = (size_t)(3 * NPTS + NPTS + 1024) * 4 + 1024 * 4;  // 270336 B
    fps_kernel<<<BATCH, 1024, fpsSmem, stream>>>(xyz, fpsIdx);
    {
        int n = BATCH * SPTS * 3;
        gather_newxyz_kernel<<<(n + 255) / 256, 256, 0, stream>>>(xyz, fpsIdx, newxyz, outXyz);
    }

    // ---- per-scale pipeline --------------------------------------------
    for (int i = 0; i < 3; ++i) {
        const int K = KS[i];
        const int Rows = BATCH * SPTS * K;

        ball_query_kernel<<<(BATCH * SPTS) / 8, 256, 0, stream>>>(xyz, newxyz, gidx, R2[i], K);
        group_kernel<<<(Rows * 64 + 255) / 256, 256, 0, stream>>>(xyz, pts, gidx, newxyz, Xbuf, Rows, K);

        int Cin = 64;
        for (int j = 0; j < 3; ++j) {
            const int Cout = MLPS[i][j];
            const float* Wp  = (const float*)d_in[2 + (i * 3 + j) * 4 + 0];
            const float* Bp  = (const float*)d_in[2 + (i * 3 + j) * 4 + 1];
            const float* Gp  = (const float*)d_in[2 + (i * 3 + j) * 4 + 2];
            const float* BTp = (const float*)d_in[2 + (i * 3 + j) * 4 + 3];

            wprep_kernel<<<(Cin * Cout + 255) / 256, 256, 0, stream>>>(Wp, wfrag, Cin, Cout);

            int groups = (Rows / 16) * (Cout / 64);
            size_t wsmem = (size_t)Cin * Cout * sizeof(bf16);   // <= 64KB
            gemm_bf16_kernel<<<(groups + 7) / 8, 256, wsmem, stream>>>(
                Xbuf, wfrag, Bp, Ybuf, Rows, Cin, Cout);

            zero_kernel<<<2, 256, 0, stream>>>(bnSum, 256);
            zero_kernel<<<2, 256, 0, stream>>>(bnSq, 256);
            stats_kernel<<<512, 256, 0, stream>>>(Ybuf, (size_t)Rows * Cout, Cout, bnSum, bnSq);
            bn_finalize_kernel<<<1, 256, 0, stream>>>(bnSum, bnSq, Gp, BTp,
                                                      1.0f / (float)Rows, bnA, bnC, Cout);
            if (j < 2) {
                size_t total = (size_t)Rows * Cout;
                bn_relu_kernel<<<(unsigned)((total + 255) / 256), 256, 0, stream>>>(
                    Ybuf, bnA, bnC, Xbuf, total, Cout);
                Cin = Cout;
            } else {
                int n = BATCH * SPTS * Cout;
                bn_relu_maxpool_kernel<<<(n + 255) / 256, 256, 0, stream>>>(
                    Ybuf, bnA, bnC, outPts, K, Cout, CHOFF[i]);
            }
        }
    }
}
